// NonLocal_17343077941864
// MI455X (gfx1250) — compile-verified, matrix operand order checked by reading, not run on hardware
//
#include <hip/hip_runtime.h>

// ---------------------------------------------------------------------------
// CDNA5 (gfx1250) non-local attention block, f16 WMMA with fp32 accumulate.
//
//   theta/phi/g = relu(W x + b)        (f16 scratch)
//   scores      = theta^T phi          (f16, per batch [T,T])
//   attn        = softmax(scores)      (in-place, fp32 math)
//   feat        = attn @ g^T           (f16, [T,C])
//   out         = relu(Ww feat + bw) + x   (fp32)
//
// All matrix math uses V_WMMA_F32_16X16X32_F16. LDS tiles are double-buffered
// (one barrier / k-step, next tile streamed while current tile is in WMMA).
// Pure-f16 tiles use GLOBAL_LOAD_ASYNC_TO_LDS_B128 (ASYNCcnt) when available.
// Workspace use: 4*B*T*C*2 + B*T*T*2 = 128 MiB.
// ---------------------------------------------------------------------------

typedef __attribute__((ext_vector_type(16))) _Float16 v16h;
typedef __attribute__((ext_vector_type(8)))  float    v8f;

#define BM 128
#define BN 128
#define BK 32
#define LDS_STRIDE 40   // BK + 8 halfs pad: 80B rows, 16B aligned, conflict-free
#define TILE_HALFS (BM * LDS_STRIDE)
#define THREADS 256

#if defined(__has_builtin)
#if __has_builtin(__builtin_amdgcn_global_load_async_to_lds_b128)
#define HAVE_ASYNC_LDS 1
#endif
#endif
#ifndef HAVE_ASYNC_LDS
#define HAVE_ASYNC_LDS 0
#endif

enum { MODE_F32_ROW = 0, MODE_F32_COL = 1, MODE_F16_ROW = 2 };
enum { EPI_RELU_H = 0, EPI_PLAIN_H = 1, EPI_RELU_RES_F = 2 };

union FragH { v16h f; unsigned int u[8]; };
union Pack4 { _Float16 h[4]; unsigned long long u; };

#if HAVE_ASYNC_LDS
// Builtin prototype (from compiler diagnostic): generic ptr-to-int4vec for the
// global side, addrspace(3) ptr-to-int4vec for the LDS side, imm offset, cpol.
typedef int v4i_ __attribute__((vector_size(16)));
typedef __attribute__((address_space(3))) v4i_* lds_v4i_ptr;
#endif

// Stage a 128 x 32 tile into LDS as f16, rows = output-dim slice, cols = K.
//  MODE_F32_ROW: src row-major [R x K] fp32, convert through VGPRs
//  MODE_F32_COL: src [K x R] fp32 (k-major), transpose + convert
//  MODE_F16_ROW: src row-major [R x K] f16 -> async DMA straight to LDS
template<int MODE>
__device__ inline void loadTile(_Float16* dst, const void* src, int ld,
                                int base, int k0, int tid) {
  if constexpr (MODE == MODE_F16_ROW) {
#if HAVE_ASYNC_LDS
    // 128 rows x 64B; 512 16B chunks over 256 lanes -> 2 async B128 per lane.
#pragma unroll
    for (int i = 0; i < 2; ++i) {
      const int c   = tid + THREADS * i;
      const int row = c >> 2;
      const int col = c & 3;  // 16B chunk within row
      const _Float16* g = (const _Float16*)src + (long)(base + row) * ld + k0 + col * 8;
      _Float16* l = dst + row * LDS_STRIDE + col * 8;
      __builtin_amdgcn_global_load_async_to_lds_b128(
          (v4i_*)g, (lds_v4i_ptr)l, 0, 0);
    }
#else
    const int tc = tid & 7;    // k-chunk of 4
    const int tr = tid >> 3;   // row 0..31
#pragma unroll
    for (int rr = 0; rr < 4; ++rr) {
      const int r = tr + 32 * rr;
      Pack4 p;
      p.u = *(const unsigned long long*)((const _Float16*)src +
                                         (long)(base + r) * ld + k0 + tc * 4);
      *(unsigned long long*)(dst + r * LDS_STRIDE + tc * 4) = p.u;
    }
#endif
  } else if constexpr (MODE == MODE_F32_COL) {
    const float* S = (const float*)src;
    const int tn = tid & 31;   // row-chunk of 4 (contiguous in source)
    const int tk = tid >> 5;   // k row 0..7
#pragma unroll
    for (int rr = 0; rr < 4; ++rr) {
      const int k = tk + 8 * rr;
      const float4 v = *(const float4*)(S + (long)(k0 + k) * ld + base + tn * 4);
      const int r4 = tn * 4;
      dst[(r4 + 0) * LDS_STRIDE + k] = (_Float16)v.x;
      dst[(r4 + 1) * LDS_STRIDE + k] = (_Float16)v.y;
      dst[(r4 + 2) * LDS_STRIDE + k] = (_Float16)v.z;
      dst[(r4 + 3) * LDS_STRIDE + k] = (_Float16)v.w;
    }
  } else {  // MODE_F32_ROW
    const int tc = tid & 7;
    const int tr = tid >> 3;
#pragma unroll
    for (int rr = 0; rr < 4; ++rr) {
      const int r = tr + 32 * rr;
      const float4 v =
          *(const float4*)((const float*)src + (long)(base + r) * ld + k0 + tc * 4);
      Pack4 p;
      p.h[0] = (_Float16)v.x; p.h[1] = (_Float16)v.y;
      p.h[2] = (_Float16)v.z; p.h[3] = (_Float16)v.w;
      *(unsigned long long*)(dst + r * LDS_STRIDE + tc * 4) = p.u;
    }
  }
}

template<int AMODE, int BMODE>
__device__ inline void asyncFence() {
#if HAVE_ASYNC_LDS
  if constexpr (AMODE == MODE_F16_ROW || BMODE == MODE_F16_ROW)
    asm volatile("s_wait_asynccnt 0x0" ::: "memory");
#endif
}

template<int AMODE, int BMODE, int EPI, bool BIASN>
__global__ __launch_bounds__(THREADS)
void gemm_wmma_kernel(const void* __restrict__ Aptr, int lda, long aStride,
                      const void* __restrict__ Bptr, int ldb, long bStride,
                      const float* __restrict__ bias,
                      const float* __restrict__ resid, int ldr, long rStride,
                      void* __restrict__ outPtr, int ldo, long oStride,
                      int K) {
  __shared__ _Float16 As[2][TILE_HALFS];
  __shared__ _Float16 Bs[2][TILE_HALFS];

  const int bz = blockIdx.z;
  const void* Ab = (AMODE == MODE_F16_ROW)
      ? (const void*)((const _Float16*)Aptr + (long)bz * aStride)
      : (const void*)((const float*)Aptr + (long)bz * aStride);
  const void* Bb = (BMODE == MODE_F16_ROW)
      ? (const void*)((const _Float16*)Bptr + (long)bz * bStride)
      : (const void*)((const float*)Bptr + (long)bz * bStride);

  const int tid   = threadIdx.x;
  const int lane  = tid & 31;
  const int wave  = tid >> 5;
  const int waveM = (wave & 3) * 32;   // 4 waves tile M
  const int waveN = (wave >> 2) * 64;  // 2 waves tile N
  const int h = lane >> 4;             // half-wave (K/N split per ISA layout)
  const int r = lane & 15;
  const int mBase = blockIdx.y * BM;
  const int nBase = blockIdx.x * BN;

  v8f acc[2][4] = {};

  const int nk = K / BK;
  loadTile<AMODE>(As[0], Ab, lda, mBase, 0, tid);
  loadTile<BMODE>(Bs[0], Bb, ldb, nBase, 0, tid);
  asyncFence<AMODE, BMODE>();
  __syncthreads();

  for (int it = 0; it < nk; ++it) {
    const int cur = it & 1;
    // Stream next k-tile into the other buffer while this tile is in WMMA.
    if (it + 1 < nk) {
      loadTile<AMODE>(As[cur ^ 1], Ab, lda, mBase, (it + 1) * BK, tid);
      loadTile<BMODE>(Bs[cur ^ 1], Bb, ldb, nBase, (it + 1) * BK, tid);
    }

    // Fragment K layout per ISA 16-bit A 16x32: vgpr v, half h ->
    //   K = (v<4?0:16) + 8*h + 2*(v&3)  (pair packed in one dword)
    const _Float16* Ac = As[cur];
    const _Float16* Bc = Bs[cur];
    FragH a[2], b[4];
#pragma unroll
    for (int v = 0; v < 8; ++v) {
      const int kk = ((v & 4) << 2) | (h << 3) | ((v & 3) << 1);
#pragma unroll
      for (int mi = 0; mi < 2; ++mi)
        a[mi].u[v] =
            *(const unsigned int*)(Ac + (waveM + mi * 16 + r) * LDS_STRIDE + kk);
#pragma unroll
      for (int ni = 0; ni < 4; ++ni)
        b[ni].u[v] =
            *(const unsigned int*)(Bc + (waveN + ni * 16 + r) * LDS_STRIDE + kk);
    }

#pragma unroll
    for (int mi = 0; mi < 2; ++mi)
#pragma unroll
      for (int ni = 0; ni < 4; ++ni)
        acc[mi][ni] = __builtin_amdgcn_wmma_f32_16x16x32_f16(
            false, a[mi].f, false, b[ni].f, (short)0, acc[mi][ni], false, false);

    asyncFence<AMODE, BMODE>();
    __syncthreads();
  }

  // Epilogue: C/D layout -> M = v + 8*h, N = lane%16 within each 16x16 tile.
#pragma unroll
  for (int mi = 0; mi < 2; ++mi) {
#pragma unroll
    for (int ni = 0; ni < 4; ++ni) {
      const int n = nBase + waveN + ni * 16 + r;
#pragma unroll
      for (int v = 0; v < 8; ++v) {
        const int m = mBase + waveM + mi * 16 + v + 8 * h;
        float val = acc[mi][ni][v];
        if constexpr (EPI != EPI_PLAIN_H) {
          val += BIASN ? bias[n] : bias[m];
          val = fmaxf(val, 0.0f);
        }
        if constexpr (EPI == EPI_RELU_RES_F) {
          const float* rb = resid + (long)bz * rStride;
          val += rb[(long)m * ldr + n];
          float* ob = (float*)outPtr + (long)bz * oStride;
          ob[(long)m * ldo + n] = val;
        } else {
          _Float16* ob = (_Float16*)outPtr + (long)bz * oStride;
          ob[(long)m * ldo + n] = (_Float16)val;
        }
      }
    }
  }
}

// In-place row softmax over f16 scores. One 256-thread block per row of 2048.
__global__ __launch_bounds__(256)
void softmax_rows_f16(_Float16* __restrict__ s, int rowLen) {
  const long row = blockIdx.x;
  _Float16* p = s + row * (long)rowLen;
  const int tid = threadIdx.x;

  float v[8];
#pragma unroll
  for (int j = 0; j < 8; ++j) v[j] = (float)p[tid + 256 * j];

  float m = v[0];
#pragma unroll
  for (int j = 1; j < 8; ++j) m = fmaxf(m, v[j]);
#pragma unroll
  for (int off = 16; off > 0; off >>= 1) m = fmaxf(m, __shfl_xor(m, off, 32));

  __shared__ float red[8];
  if ((tid & 31) == 0) red[tid >> 5] = m;
  __syncthreads();
  float gmax = red[0];
#pragma unroll
  for (int i = 1; i < 8; ++i) gmax = fmaxf(gmax, red[i]);
  __syncthreads();

  float sum = 0.0f;
#pragma unroll
  for (int j = 0; j < 8; ++j) { v[j] = __expf(v[j] - gmax); sum += v[j]; }
#pragma unroll
  for (int off = 16; off > 0; off >>= 1) sum += __shfl_xor(sum, off, 32);
  if ((tid & 31) == 0) red[tid >> 5] = sum;
  __syncthreads();
  float tot = 0.0f;
#pragma unroll
  for (int i = 0; i < 8; ++i) tot += red[i];
  const float inv = 1.0f / tot;
#pragma unroll
  for (int j = 0; j < 8; ++j) p[tid + 256 * j] = (_Float16)(v[j] * inv);
}

extern "C" void kernel_launch(void* const* d_in, const int* in_sizes, int n_in,
                              void* d_out, int out_size, void* d_ws, size_t ws_size,
                              hipStream_t stream) {
  (void)in_sizes; (void)n_in; (void)out_size; (void)ws_size;
  const float* x       = (const float*)d_in[0];
  const float* w_theta = (const float*)d_in[1];
  const float* b_theta = (const float*)d_in[2];
  const float* w_phi   = (const float*)d_in[3];
  const float* b_phi   = (const float*)d_in[4];
  const float* w_g     = (const float*)d_in[5];
  const float* b_g     = (const float*)d_in[6];
  const float* w_w     = (const float*)d_in[7];
  const float* b_w     = (const float*)d_in[8];

  const int B = 8, C = 512, T = 2048;
  const long TC = (long)T * C;
  const long TT = (long)T * T;
  const long CT = (long)C * T;

  // Workspace layout (f16): thetaT/phiT/featT are [B][T][C]; g is [B][C][T];
  // attn is [B][T][T] (scores computed here, softmaxed in place). 128 MiB.
  _Float16* thetaT = (_Float16*)d_ws;
  _Float16* phiT   = thetaT + (long)B * TC;
  _Float16* gbuf   = phiT   + (long)B * TC;
  _Float16* featT  = gbuf   + (long)B * TC;
  _Float16* attn   = featT  + (long)B * TC;

  dim3 blk(THREADS);

  // thetaT[t][o] = relu(sum_c x[c][t] * Wt[o][c] + bt[o])   (coalesced stores)
  gemm_wmma_kernel<MODE_F32_COL, MODE_F32_ROW, EPI_RELU_H, true>
      <<<dim3(C / BN, T / BM, B), blk, 0, stream>>>(
      x, T, CT, w_theta, C, 0, b_theta, nullptr, 0, 0, thetaT, C, TC, C);
  gemm_wmma_kernel<MODE_F32_COL, MODE_F32_ROW, EPI_RELU_H, true>
      <<<dim3(C / BN, T / BM, B), blk, 0, stream>>>(
      x, T, CT, w_phi, C, 0, b_phi, nullptr, 0, 0, phiT, C, TC, C);

  // g[o][t] = relu(sum_c Wg[o][c] * x[c][t] + bg[o])
  gemm_wmma_kernel<MODE_F32_ROW, MODE_F32_COL, EPI_RELU_H, false>
      <<<dim3(T / BN, C / BM, B), blk, 0, stream>>>(
      w_g, C, 0, x, T, CT, b_g, nullptr, 0, 0, gbuf, T, CT, C);

  // scores[i][j] = sum_c thetaT[i][c] * phiT[j][c]   -> attn buffer
  gemm_wmma_kernel<MODE_F16_ROW, MODE_F16_ROW, EPI_PLAIN_H, false>
      <<<dim3(T / BN, T / BM, B), blk, 0, stream>>>(
      thetaT, C, TC, phiT, C, TC, nullptr, nullptr, 0, 0, attn, T, TT, C);

  // attn = softmax(scores) in place, per row
  softmax_rows_f16<<<dim3(B * T), blk, 0, stream>>>(attn, T);

  // featT[i][c] = sum_j attn[i][j] * g[c][j]
  gemm_wmma_kernel<MODE_F16_ROW, MODE_F16_ROW, EPI_PLAIN_H, false>
      <<<dim3(C / BN, T / BM, B), blk, 0, stream>>>(
      attn, T, TT, gbuf, T, CT, nullptr, nullptr, 0, 0, featT, C, TC, T);

  // out[o][t] = relu(sum_c Ww[o][c] * featT[t][c] + bw[o]) + x[o][t]
  gemm_wmma_kernel<MODE_F32_ROW, MODE_F16_ROW, EPI_RELU_RES_F, false>
      <<<dim3(T / BN, C / BM, B), blk, 0, stream>>>(
      w_w, C, 0, featT, C, TC, b_w, x, T, CT, d_out, T, CT, C);
}
